// W8Linear_54649163874458
// MI455X (gfx1250) — matrix operand order repaired and checked by reading, not compiled, not run on hardware
//
#include <hip/hip_runtime.h>
#include <hip/hip_bf16.h>

// W8Linear: out[M,N] = x[M,K] @ dequant(Wq[N,K]).T + bias, bf16 in/out.
// M = B*S = 8192, K = 4096, N = 11008. Compute-bound bf16 WMMA GEMM.
// Double-buffered LDS; A tiles staged by the Tensor Data Mover (TDM),
// W tiles staged through VALU int8->bf16 dequant that co-executes with WMMA.

typedef __attribute__((ext_vector_type(16))) __bf16 v16bf;
typedef __attribute__((ext_vector_type(8)))  float  v8f;
typedef __attribute__((ext_vector_type(4)))  unsigned v4u;
typedef __attribute__((ext_vector_type(8)))  int      v8i;
typedef __attribute__((ext_vector_type(4)))  int      v4i;

#define M_TOT 8192
#define K_TOT 4096
#define N_TOT 11008

#define BM 128
#define BN 128
#define BK 64
#define KT (K_TOT / BK)   // 64 pipelined k-steps
#define LDSS 72           // padded LDS row stride: 144B, 16B-aligned, bank-conflict-free

#if defined(__has_builtin)
#  if __has_builtin(__builtin_amdgcn_tensor_load_to_lds) && \
      __has_builtin(__builtin_amdgcn_s_wait_tensorcnt)
#    define HAVE_TDM 1
#  endif
#endif
#ifndef HAVE_TDM
#  define HAVE_TDM 0
#endif

static __device__ __forceinline__ unsigned short f2bf(float f) {
  unsigned u = __float_as_uint(f);
  u += 0x7FFFu + ((u >> 16) & 1u);      // round-to-nearest-even
  return (unsigned short)(u >> 16);
}
static __device__ __forceinline__ float bf2f(unsigned short h) {
  return __uint_as_float(((unsigned)h) << 16);
}
static __device__ __forceinline__ unsigned pack_bf16(float f0, float f1) {
  union { __hip_bfloat162 h2; unsigned u; } c;
  c.h2 = __float22bfloat162_rn(make_float2(f0, f1));   // -> v_cvt_pk_bf16_f32
  return c.u;
}

// 16 int8 -> 16 bf16 with w' = w*s + c  (c = -zp*s folds the zero-point)
static __device__ __forceinline__ void dequant16(int4 wq, float s, float c,
                                                 unsigned out[8]) {
  const int wv[4] = {wq.x, wq.y, wq.z, wq.w};
#pragma unroll
  for (int j = 0; j < 4; ++j) {
    int v = wv[j];
    float f0 = fmaf((float)(signed char)(v),       s, c);
    float f1 = fmaf((float)(signed char)(v >> 8),  s, c);
    float f2 = fmaf((float)(signed char)(v >> 16), s, c);
    float f3 = fmaf((float)(v >> 24),              s, c);
    out[2 * j]     = pack_bf16(f0, f1);
    out[2 * j + 1] = pack_bf16(f2, f3);
  }
}

#if HAVE_TDM
// TDM descriptor: 2-D tile, 128 rows x 64 bf16 elems, row stride K_TOT elems,
// LDS padding 4 DWORDs (16B) after every 32 DWORDs (one 128B row) -> 144B stride.
static __device__ __forceinline__ void tdm_load_A(const unsigned short* gsrc,
                                                  unsigned lds_off) {
  unsigned long long ga = (unsigned long long)(uintptr_t)gsrc;
  v4u g0;
  g0[0] = 1u;                                            // count = 1 valid descriptor
  g0[1] = lds_off;                                       // lds_addr (bytes)
  g0[2] = (unsigned)(ga & 0xFFFFFFFFu);                  // global_addr[31:0]
  g0[3] = (unsigned)((ga >> 32) & 0x01FFFFFFu)           // global_addr[56:32]
        | (2u << 30);                                    // type = 2 ("image")
  v8i g1;
  g1[0] = (int)((1u << 16)      // data_size = 1 -> 2 bytes
              | (1u << 20)      // pad_enable
              | (4u << 22)      // pad_interval code 4 -> every 32 DWORDs
              | (3u << 25));    // pad_amount  code 3 -> 4 DWORDs (16B)
  g1[1] = (int)((unsigned)K_TOT << 16);   // tensor_dim0[15:0] (=4096) in bits[63:48]
  g1[2] = (int)(128u << 16);              // tensor_dim0[31:16]=0, tensor_dim1=128
  g1[3] = (int)(64u << 16);               // tensor_dim1[31:16]=0, tile_dim0=64
  g1[4] = (int)128;                       // tile_dim1 = 128, tile_dim2 = 0
  g1[5] = (int)K_TOT;                     // tensor_dim0_stride[31:0] = 4096
  g1[6] = 0;                              // stride[47:32]=0, tensor_dim1_stride lo=0
  g1[7] = 0;
  v4i g2 = {0, 0, 0, 0};                  // dims 2/3 unused
  v4i g3 = {0, 0, 0, 0};
#if defined(__clang_major__) && (__clang_major__ >= 23)
  v8i g4 = {0, 0, 0, 0, 0, 0, 0, 0};
  __builtin_amdgcn_tensor_load_to_lds(g0, g1, g2, g3, g4, 0);
#else
  __builtin_amdgcn_tensor_load_to_lds(g0, g1, g2, g3, 0);
#endif
}
#endif  // HAVE_TDM

union FragBF16 { uint4 q[2]; v16bf v; };

__global__ __launch_bounds__(256)
void w8linear_wmma_kernel(const unsigned short* __restrict__ X,     // bf16 raw [M,K]
                          const signed char*   __restrict__ Wq,     // int8 [N,K]
                          const unsigned short* __restrict__ Scale, // bf16 raw [N]
                          const int*           __restrict__ ZP,     // scalar
                          const unsigned short* __restrict__ Bias,  // bf16 raw [N]
                          unsigned short*      __restrict__ Out) {  // bf16 raw [M,N]
  __shared__ __align__(16) unsigned short As[2][BM][LDSS];
  __shared__ __align__(16) unsigned short Bs[2][BN][LDSS];

  const int tid  = threadIdx.x;       // 0..255
  const int lane = tid & 31;
  const int wv   = tid >> 5;          // wave 0..7
  const int wm   = wv >> 2;           // 0..1 : 64-row slab of C
  const int wn   = wv & 3;            // 0..3 : 32-col slab of C

  const int m0 = blockIdx.y * BM;
  const int n0 = blockIdx.x * BN;

  // ---- fixed per-thread loader roles (weights) ----
  const int rowW = tid >> 2;          // 0..63  (handles rows rowW + 64*i, i=0..1)
  const int segW = tid & 3;           // 16B (16 x int8) chunk within 64B row

  const float zpf = (float)ZP[0];
  float sc[2], cc[2];
#pragma unroll
  for (int i = 0; i < 2; ++i) {
    sc[i] = bf2f(Scale[n0 + rowW + 64 * i]);
    cc[i] = -zpf * sc[i];
  }

  const signed char* wBase = Wq + (size_t)(n0 + rowW) * K_TOT + segW * 16;
  const unsigned short* xTile = X + (size_t)m0 * K_TOT;     // A tile col 0

#if !HAVE_TDM
  const int rowA = tid >> 3;          // 0..31  (handles rows rowA + 32*i, i=0..3)
  const int segA = tid & 7;           // 16B chunk within 128B row
  const unsigned short* xBase = X + (size_t)(m0 + rowA) * K_TOT + segA * 8;
#endif

  auto loadW = [&](int k0, int4 rw[2]) {
#pragma unroll
    for (int i = 0; i < 2; ++i)
      rw[i] = *reinterpret_cast<const int4*>(wBase + (size_t)(64 * i) * K_TOT + k0);
  };
  auto storeW = [&](int p, const int4 rw[2]) {
#pragma unroll
    for (int i = 0; i < 2; ++i) {
      unsigned wd[8];
      dequant16(rw[i], sc[i], cc[i], wd);
      *reinterpret_cast<uint4*>(&Bs[p][rowW + 64 * i][segW * 16]) =
          make_uint4(wd[0], wd[1], wd[2], wd[3]);
      *reinterpret_cast<uint4*>(&Bs[p][rowW + 64 * i][segW * 16 + 8]) =
          make_uint4(wd[4], wd[5], wd[6], wd[7]);
    }
  };
#if !HAVE_TDM
  auto loadA = [&](int k0, uint4 ra[4]) {
#pragma unroll
    for (int i = 0; i < 4; ++i)
      ra[i] = *reinterpret_cast<const uint4*>(xBase + (size_t)(32 * i) * K_TOT + k0);
  };
  auto storeA = [&](int p, const uint4 ra[4]) {
#pragma unroll
    for (int i = 0; i < 4; ++i)
      *reinterpret_cast<uint4*>(&As[p][rowA + 32 * i][segA * 8]) = ra[i];
  };
#endif

  v8f acc[4][2];
#pragma unroll
  for (int mt = 0; mt < 4; ++mt)
#pragma unroll
    for (int nt = 0; nt < 2; ++nt)
#pragma unroll
      for (int r = 0; r < 8; ++r) acc[mt][nt][r] = 0.0f;

  const int am  = lane & 15;          // row/col within 16x16 tile
  const int kba = (lane >> 4) * 8;    // A frag: K chunk select
  const int kbb = (lane >> 4) * 8;    // B frag: K chunk select

  // ---- prologue: stage tile 0 into buffer 0 ----
  {
    int4 rw[2];
    loadW(0, rw);
    storeW(0, rw);
#if HAVE_TDM
    if (wv == 0) {
      tdm_load_A(xTile, (unsigned)(uintptr_t)&As[0][0][0]);
      __builtin_amdgcn_s_wait_tensorcnt(0);
    }
#else
    uint4 ra[4];
    loadA(0, ra);
    storeA(0, ra);
#endif
  }
  __syncthreads();

  // ---- main pipeline: 1 barrier / 64-K step; TDM + global loads hide under WMMA ----
  int4 rw[2];
#if !HAVE_TDM
  uint4 ra[4];
#endif
  for (int kt = 0; kt < KT; ++kt) {
    const int p = kt & 1;
    const bool more = (kt + 1) < KT;
    if (more) {
#if HAVE_TDM
      if (wv == 0)
        tdm_load_A(xTile + (kt + 1) * BK,
                   (unsigned)(uintptr_t)&As[p ^ 1][0][0]);   // DMA into back buffer
#else
      loadA((kt + 1) * BK, ra);
#endif
      loadW((kt + 1) * BK, rw);
    }
    if (kt + 2 < KT)
      __builtin_prefetch(wBase + (kt + 2) * BK, 0, 0);       // global_prefetch_b8

#pragma unroll
    for (int kk = 0; kk < 2; ++kk) {
      const int ko = kk * 32;
      FragBF16 a[4], b[2];
#pragma unroll
      for (int mt = 0; mt < 4; ++mt) {
        const unsigned short* pa = &As[p][wm * 64 + mt * 16 + am][ko];
        a[mt].q[0] = *reinterpret_cast<const uint4*>(pa + kba);       // K 0..7 / 8..15
        a[mt].q[1] = *reinterpret_cast<const uint4*>(pa + 16 + kba);  // K 16..23 / 24..31
      }
#pragma unroll
      for (int nt = 0; nt < 2; ++nt) {
        const unsigned short* pb = &Bs[p][wn * 32 + nt * 16 + am][ko + kbb];
        b[nt].q[0] = *reinterpret_cast<const uint4*>(pb);             // K 0..7  / 8..15
        b[nt].q[1] = *reinterpret_cast<const uint4*>(pb + 16);        // K 16..23 / 24..31
      }
#pragma unroll
      for (int mt = 0; mt < 4; ++mt)
#pragma unroll
        for (int nt = 0; nt < 2; ++nt)
          acc[mt][nt] = __builtin_amdgcn_wmma_f32_16x16x32_bf16(
              false, a[mt].v, false, b[nt].v, (short)0, acc[mt][nt], false, false);
    }

    if (more) {
      storeW(p ^ 1, rw);
#if !HAVE_TDM
      storeA(p ^ 1, ra);
#endif
    }
#if HAVE_TDM
    if (wv == 0) __builtin_amdgcn_s_wait_tensorcnt(0);  // A DMA done before barrier
#endif
    __syncthreads();
  }

  // ---- epilogue: bias + bf16 store (C layout: N=lane&15, M=8*(lane>=16)+vgpr) ----
  const int cr = (lane >> 4) * 8;
#pragma unroll
  for (int mt = 0; mt < 4; ++mt) {
    const int gr = m0 + wm * 64 + mt * 16 + cr;
#pragma unroll
    for (int nt = 0; nt < 2; ++nt) {
      const int gc = n0 + wn * 32 + nt * 16 + am;
      const float bsf = bf2f(Bias[gc]);
      unsigned short* o = Out + (size_t)gr * N_TOT + gc;
#pragma unroll
      for (int r = 0; r < 8; ++r)
        o[(size_t)r * N_TOT] = f2bf(acc[mt][nt][r] + bsf);
    }
  }
}

extern "C" void kernel_launch(void* const* d_in, const int* in_sizes, int n_in,
                              void* d_out, int out_size, void* d_ws, size_t ws_size,
                              hipStream_t stream) {
  (void)in_sizes; (void)n_in; (void)out_size; (void)d_ws; (void)ws_size;
  const unsigned short* X     = (const unsigned short*)d_in[0]; // bf16 [4,2048,4096]
  const signed char*    Wq    = (const signed char*)   d_in[1]; // int8 [11008,4096]
  const unsigned short* Scale = (const unsigned short*)d_in[2]; // bf16 [11008,1]
  const int*            ZP    = (const int*)           d_in[3]; // scalar
  const unsigned short* Bias  = (const unsigned short*)d_in[4]; // bf16 [11008]
  unsigned short*       Out   = (unsigned short*)      d_out;   // bf16 [4,2048,11008]

  dim3 grid(N_TOT / BN, M_TOT / BM);  // (86, 64)
  dim3 block(256);
  w8linear_wmma_kernel<<<grid, block, 0, stream>>>(X, Wq, Scale, ZP, Bias, Out);
}
